// LlamaAttention_80487687127348
// MI455X (gfx1250) — compile-verified, matrix operand order checked by reading, not run on hardware
//
#include <hip/hip_runtime.h>
#include <hip/hip_bf16.h>

typedef float  v8f   __attribute__((ext_vector_type(8)));
typedef __bf16 v16bf __attribute__((ext_vector_type(16)));
typedef __bf16 v8bf  __attribute__((ext_vector_type(8)));
typedef int    v4i   __attribute__((vector_size(16)));  // b128 payload type

#define B_  2
#define S_  2048
#define D_  2048
#define H_  16
#define HD_ 128

#define AS1 __attribute__((address_space(1)))
#define AS3 __attribute__((address_space(3)))

// ---------------------------------------------------------------------------
// CDNA5 async global->LDS staging (ASYNCcnt-tracked), guarded so that a
// toolchain without the builtins falls back to synchronous staging.
// Signature (confirmed by clang diagnostics): (v4i AS1*, v4i AS3*, imm, imm).
// ---------------------------------------------------------------------------
#if __has_builtin(__builtin_amdgcn_global_load_async_to_lds_b128) && \
    __has_builtin(__builtin_amdgcn_s_wait_asynccnt)
#define USE_ASYNC_LDS 1
#else
#define USE_ASYNC_LDS 0
#endif

#if USE_ASYNC_LDS
// stage 16 contiguous bf16 (2 x b128) from global to LDS, no VGPR data path
#define ASYNC_STAGE16(l, g)                                                  \
    do {                                                                     \
        __builtin_amdgcn_global_load_async_to_lds_b128(                      \
            (AS1 v4i*)(g), (AS3 v4i*)(l), 0, 0);                             \
        __builtin_amdgcn_global_load_async_to_lds_b128(                      \
            (AS1 v4i*)((g) + 8), (AS3 v4i*)((l) + 8), 0, 0);                 \
    } while (0)
#define WAIT_ASYNC(n) __builtin_amdgcn_s_wait_asynccnt(n)
#else
#define ASYNC_STAGE16(l, g)                                                  \
    do {                                                                     \
        *(v8bf*)(l)       = *(const v8bf*)(g);                               \
        *(v8bf*)((l) + 8) = *(const v8bf*)((g) + 8);                         \
    } while (0)
#define WAIT_ASYNC(n)
#endif

// ---------------------------------------------------------------------------
// CDNA5 Tensor Data Mover (TENSORcnt-tracked).  Arity differs by toolchain:
// ROCm 7.2 clang: 5 args; amdgpu-toolchain (ships amd_gfx1250_TDM.h): 6 args.
// ---------------------------------------------------------------------------
#if __has_builtin(__builtin_amdgcn_tensor_load_to_lds) && \
    __has_builtin(__builtin_amdgcn_s_wait_tensorcnt)
#define USE_TDM 1
#else
#define USE_TDM 0
#endif

#if __has_include(<hip/amd_detail/amd_gfx1250_TDM.h>)
#define TDM_SIX_ARGS 1
#else
#define TDM_SIX_ARGS 0
#endif

#if USE_TDM
typedef unsigned int u32x4 __attribute__((ext_vector_type(4)));
typedef int          i32x8 __attribute__((ext_vector_type(8)));
typedef int          i32x4 __attribute__((ext_vector_type(4)));

// One-shot 2D tile DMA: global (strided rows) -> LDS (optionally padded rows).
// D# packing per CDNA5 ISA 8.3/8.4 (group0: count/lds/global/type,
// group1: data_size, pad, dims, strides; groups 2-3 zero for 2D tiles).
static __device__ __forceinline__ void tdm_load_2d_bf16(
    const __bf16* gsrc, unsigned lds_byte_off,
    unsigned tensor_w, unsigned tensor_h, unsigned stride_elems,
    unsigned tile_w, unsigned tile_h,
    unsigned pad_interval_code, unsigned pad_amount_code, bool pad_en) {
    const unsigned long long ga = (unsigned long long)(uintptr_t)gsrc;
    u32x4 g0;
    g0[0] = 1u;                                   // count=1 (valid user D#)
    g0[1] = lds_byte_off;                         // lds_addr
    g0[2] = (unsigned)(ga & 0xFFFFFFFFu);         // global_addr[31:0]
    g0[3] = (unsigned)((ga >> 32) & 0x01FFFFFFu)  // global_addr[56:32]
            | (2u << 30);                         // type=2 ("image")
    i32x8 g1;
    g1[0] = (int)((1u << 16)                      // data_size = 1 (2 bytes)
                  | (pad_en ? (1u << 20) : 0u)    // pad_enable
                  | (pad_interval_code << 22)
                  | (pad_amount_code << 25));
    g1[1] = (int)((tensor_w & 0xFFFFu) << 16);                      // dim0 lo
    g1[2] = (int)(((tensor_w >> 16) & 0xFFFFu) |
                  ((tensor_h & 0xFFFFu) << 16));                    // dim0 hi | dim1 lo
    g1[3] = (int)(((tensor_h >> 16) & 0xFFFFu) |
                  ((tile_w & 0xFFFFu) << 16));                      // dim1 hi | tile0
    g1[4] = (int)(tile_h & 0xFFFFu);              // tile1 | tile2(=0)
    g1[5] = (int)stride_elems;                    // tensor_dim0_stride[31:0]
    g1[6] = 0;                                    // stride0[47:32] | stride1 lo
    g1[7] = 0;
    i32x4 gz = {0, 0, 0, 0};
#if TDM_SIX_ARGS
    i32x8 gz8 = {0, 0, 0, 0, 0, 0, 0, 0};
    __builtin_amdgcn_tensor_load_to_lds(g0, g1, gz, gz, gz8, 0);
#else
    __builtin_amdgcn_tensor_load_to_lds(g0, g1, gz, gz, 0);
#endif
}
#define WAIT_TENSOR() __builtin_amdgcn_s_wait_tensorcnt(0)
#endif  // USE_TDM

static __device__ __forceinline__ v8f zero8() {
    v8f z = {0.f,0.f,0.f,0.f,0.f,0.f,0.f,0.f};
    return z;
}

// Load a 16-lane-striped 16-bit WMMA operand fragment.
// `row` points to 32 contiguous bf16 (the K slice for this lane's row/col).
// Per ISA layout: lanes 0-15 hold K in [0,8)+[16,24), lanes 16-31 [8,16)+[24,32).
static __device__ __forceinline__ v16bf load_frag(const __bf16* row, int hi) {
    v8bf a = *(const v8bf*)(row + hi * 8);
    v8bf b = *(const v8bf*)(row + 16 + hi * 8);
    return __builtin_shufflevector(a, b, 0,1,2,3,4,5,6,7,8,9,10,11,12,13,14,15);
}

static __device__ __forceinline__ v8f wmma_bf16(v16bf a, v16bf b, v8f c) {
    return __builtin_amdgcn_wmma_f32_16x16x32_bf16(
        /*neg_a=*/false, a, /*neg_b=*/false, b,
        /*c_mod=*/(short)0, c, /*reuse_a=*/false, /*reuse_b=*/false);
}

// ---------------------------------------------------------------------------
// f32 -> bf16 conversion (bandwidth-trivial pre-pass); 8 elems / thread.
// n8 = element_count / 8
// ---------------------------------------------------------------------------
__global__ __launch_bounds__(256) void cvt_f32_bf16(const float* __restrict__ in,
                                                    __bf16* __restrict__ out,
                                                    size_t n8) {
    size_t i = (size_t)blockIdx.x * blockDim.x + threadIdx.x;
    size_t stride = (size_t)gridDim.x * blockDim.x;
    for (; i < n8; i += stride) {
        const float* p = in + i * 8;
        v8bf o;
#pragma unroll
        for (int j = 0; j < 8; ++j) o[j] = (__bf16)p[j];
        *(v8bf*)(out + i * 8) = o;
    }
}

// ---------------------------------------------------------------------------
// C = A (MxK, bf16 row-major) x B^T (B is NxK bf16 row-major)  [NT GEMM]
// Block tile 128x128, 256 threads = 8 waves (4 M-waves x 2 N-waves),
// wave tile 32x64 = 2x4 WMMA 16x16 accumulators. K step 32.
// Double-buffered LDS with async global->LDS staging.
// ---------------------------------------------------------------------------
template <bool BF16OUT>
__global__ __launch_bounds__(256) void gemm_bf16_nt(const __bf16* __restrict__ A,
                                                    const __bf16* __restrict__ Bm,
                                                    float* __restrict__ Cf,
                                                    __bf16* __restrict__ Cb,
                                                    int M, int N, int K) {
    constexpr int LDSK = 40;  // padded row stride in halfs (80B, 16B aligned)
    __shared__ __bf16 sA[2][128 * LDSK];
    __shared__ __bf16 sB[2][128 * LDSK];

    const int tid  = threadIdx.x;
    const int wave = tid >> 5;
    const int lane = tid & 31;
    const int wm = wave >> 1;        // 0..3
    const int wn = wave & 1;         // 0..1
    const int m0 = blockIdx.y * 128;
    const int n0 = blockIdx.x * 128;
    const int lm = lane & 15;
    const int hi = lane >> 4;

    const int sr = tid >> 1;         // 0..127 staged row
    const int sc = (tid & 1) * 16;   // 0 or 16 (halfs)
    const __bf16* gA = A  + (size_t)(m0 + sr) * K + sc;
    const __bf16* gB = Bm + (size_t)(n0 + sr) * K + sc;
    __bf16* lA0 = &sA[0][sr * LDSK + sc];
    __bf16* lB0 = &sB[0][sr * LDSK + sc];
    __bf16* lA1 = &sA[1][sr * LDSK + sc];
    __bf16* lB1 = &sB[1][sr * LDSK + sc];

    v8f acc[2][4];
#pragma unroll
    for (int i = 0; i < 2; ++i)
#pragma unroll
        for (int j = 0; j < 4; ++j) acc[i][j] = zero8();

    // Prologue: stage tile 0 into buffer 0 (4 async ops / wave)
    ASYNC_STAGE16(lA0, gA);
    ASYNC_STAGE16(lB0, gB);

    const int T = K >> 5;
    for (int t = 0; t < T; ++t) {
        const int buf = t & 1;
        if (t + 1 < T) {
            // stage next tile into the other buffer, then wait only for the
            // current tile (async loads complete in order: 4 remain in flight)
            if (buf == 0) { ASYNC_STAGE16(lA1, gA + (t + 1) * 32);
                            ASYNC_STAGE16(lB1, gB + (t + 1) * 32); }
            else          { ASYNC_STAGE16(lA0, gA + (t + 1) * 32);
                            ASYNC_STAGE16(lB0, gB + (t + 1) * 32); }
            WAIT_ASYNC(4);
        } else {
            WAIT_ASYNC(0);
        }
        __syncthreads();

        const __bf16* cA = &sA[buf][0];
        const __bf16* cB = &sB[buf][0];
        v16bf af[2], bfr[4];
#pragma unroll
        for (int i = 0; i < 2; ++i)
            af[i] = load_frag(cA + (wm * 32 + i * 16 + lm) * LDSK, hi);
#pragma unroll
        for (int j = 0; j < 4; ++j)
            bfr[j] = load_frag(cB + (wn * 64 + j * 16 + lm) * LDSK, hi);
#pragma unroll
        for (int i = 0; i < 2; ++i)
#pragma unroll
            for (int j = 0; j < 4; ++j)
                acc[i][j] = wmma_bf16(af[i], bfr[j], acc[i][j]);
        __syncthreads();  // protect buffer reuse before next-next stage
    }

    // Epilogue: C/D layout -> lane holds N=lm, VGPR r holds M = r + 8*hi
#pragma unroll
    for (int i = 0; i < 2; ++i) {
#pragma unroll
        for (int j = 0; j < 4; ++j) {
            const int mb = m0 + wm * 32 + i * 16 + hi * 8;
            const int nb = n0 + wn * 64 + j * 16 + lm;
#pragma unroll
            for (int r = 0; r < 8; ++r) {
                size_t idx = (size_t)(mb + r) * N + nb;
                if (BF16OUT) Cb[idx] = (__bf16)acc[i][j][r];
                else         Cf[idx] = acc[i][j][r];
            }
        }
    }
}

// ---------------------------------------------------------------------------
// RoPE, in-place on bf16 Q and K.  One thread per (b, s, h, pair i of 64).
// ---------------------------------------------------------------------------
__global__ __launch_bounds__(256) void rope_kernel(__bf16* __restrict__ q,
                                                   __bf16* __restrict__ k) {
    const int idx = blockIdx.x * blockDim.x + threadIdx.x;
    const int i = idx & 63;
    const int h = (idx >> 6) & (H_ - 1);
    const int s = (idx >> 10) & (S_ - 1);
    const int b = idx >> 21;
    if (b >= B_) return;

    // inv_freq = base^(-2i/HD)
    const float inv = __expf(-((float)(2 * i) / (float)HD_) * logf(10000.0f));
    const float f = (float)s * inv;
    float sn, cs;
    __sincosf(f, &sn, &cs);

    const size_t base = ((size_t)b * S_ + s) * D_ + (size_t)h * HD_;
    {
        float x1 = (float)q[base + i], x2 = (float)q[base + i + 64];
        q[base + i]      = (__bf16)(x1 * cs - x2 * sn);
        q[base + i + 64] = (__bf16)(x2 * cs + x1 * sn);
    }
    {
        float x1 = (float)k[base + i], x2 = (float)k[base + i + 64];
        k[base + i]      = (__bf16)(x1 * cs - x2 * sn);
        k[base + i + 64] = (__bf16)(x2 * cs + x1 * sn);
    }
}

// ---------------------------------------------------------------------------
// Flash attention (causal, online softmax).  One block = (b, h, 64 q-rows),
// 128 threads = 4 waves, each wave owns 16 q rows.  KV streamed 32 rows/step.
// K chunk staged by the Tensor Data Mover (one descriptor, wave 0, padded
// rows) when available, else async global->LDS; V staged transposed.
// ---------------------------------------------------------------------------
__global__ __launch_bounds__(128) void flash_kernel(const __bf16* __restrict__ Q,
                                                    const __bf16* __restrict__ Kb,
                                                    const __bf16* __restrict__ Vb,
                                                    __bf16* __restrict__ O) {
    constexpr int KLDS = 136;  // K chunk row stride (halfs): 128 + 8 pad
    constexpr int VLDS = 40;   // V^T row stride (halfs)
    __shared__ __bf16 sK[32 * KLDS];       // 32 kv rows x 128 features (padded)
    __shared__ __bf16 sVt[128 * VLDS];     // transposed: [feature][kv]
    __shared__ __bf16 sP[4][16 * 32];      // per-wave P scratch

    const int bid = blockIdx.x;
    const int qt = bid & 31;               // S/64 = 32 q tiles
    const int h  = (bid >> 5) & (H_ - 1);
    const int b  = bid >> 9;

    const int wave = threadIdx.x >> 5;
    const int lane = threadIdx.x & 31;
    const int lm = lane & 15;
    const int hi = lane >> 4;

    const int qbase = qt * 64 + wave * 16;          // wave's first q row
    const size_t headoff = (size_t)h * HD_;
    const __bf16* qp = Q + ((size_t)b * S_) * D_ + headoff;

#if USE_TDM
    const unsigned sK_base = (unsigned)(uintptr_t)(AS3 char*)&sK[0];
#endif

    // Q fragments: 16 rows x 128 features, 4 K-steps of 32
    v16bf qf[4];
#pragma unroll
    for (int kk = 0; kk < 4; ++kk)
        qf[kk] = load_frag(qp + (size_t)(qbase + lm) * D_ + kk * 32, hi);

    v8f o[8];
#pragma unroll
    for (int t = 0; t < 8; ++t) o[t] = zero8();
    float mrow[8], lrow[8];
#pragma unroll
    for (int r = 0; r < 8; ++r) { mrow[r] = -3.0e38f; lrow[r] = 0.f; }
    const float scale = 0.08838834764831845f;  // 1/sqrt(128)

    const int kvend = qt * 64 + 64;
    for (int kv0 = 0; kv0 < kvend; kv0 += 32) {
        // ---- Stage K (TDM / async) and V (transposed), 32x128 each ----
        {
            const int r = threadIdx.x >> 2;        // 0..31
            const int c = (threadIdx.x & 3) * 32;  // 0/32/64/96
#if USE_TDM
            if (wave == 0) {
                // one descriptor: 32 rows x 128 bf16, row stride 2048 elems,
                // LDS pad 4 DWORDs (code 3) every 64 DWORDs (code 5) -> 136
                const __bf16* kt = Kb + ((size_t)(b * S_ + kv0)) * D_ + headoff;
                tdm_load_2d_bf16(kt, sK_base,
                                 /*tensor_w=*/HD_,
                                 /*tensor_h=*/(unsigned)(B_ * S_ - (b * S_ + kv0)),
                                 /*stride_elems=*/D_,
                                 /*tile_w=*/HD_, /*tile_h=*/32,
                                 /*pad_interval=*/5, /*pad_amount=*/3, true);
            }
#else
            const __bf16* kg = Kb + ((size_t)(b * S_ + kv0 + r)) * D_ + headoff + c;
            ASYNC_STAGE16(sK + r * KLDS + c,      kg);
            ASYNC_STAGE16(sK + r * KLDS + c + 16, kg + 16);
#endif
            const __bf16* vg = Vb + ((size_t)(b * S_ + kv0 + r)) * D_ + headoff + c;
            v8bf vv[4];
#pragma unroll
            for (int q8 = 0; q8 < 4; ++q8) vv[q8] = *(const v8bf*)(vg + q8 * 8);
#pragma unroll
            for (int q8 = 0; q8 < 4; ++q8)
#pragma unroll
                for (int j = 0; j < 8; ++j)
                    sVt[(c + q8 * 8 + j) * VLDS + r] = vv[q8][j];
            if (kv0 + 32 < kvend)
                __builtin_prefetch(vg + (size_t)32 * D_, 0, 1);
#if USE_TDM
            if (wave == 0) WAIT_TENSOR();
#else
            WAIT_ASYNC(0);
#endif
        }
        __syncthreads();

        // ---- S = Q K^T for two 16-col kv tiles ----
        v8f s0 = zero8(), s1 = zero8();
#pragma unroll
        for (int kk = 0; kk < 4; ++kk) {
            v16bf b0 = load_frag(sK + (lm)      * KLDS + kk * 32, hi);
            v16bf b1 = load_frag(sK + (16 + lm) * KLDS + kk * 32, hi);
            s0 = wmma_bf16(qf[kk], b0, s0);
            s1 = wmma_bf16(qf[kk], b1, s1);
        }

        // ---- scale + causal mask (lane=N col, VGPR r -> row hi*8+r) ----
        const int qrow0 = qbase + hi * 8;
#pragma unroll
        for (int r = 0; r < 8; ++r) {
            const int qr = qrow0 + r;
            float e0 = s0[r] * scale;
            float e1 = s1[r] * scale;
            if (kv0 + lm      > qr) e0 = -3.0e38f;
            if (kv0 + 16 + lm > qr) e1 = -3.0e38f;
            s0[r] = e0; s1[r] = e1;
        }

        // ---- online softmax over 32 kv cols ----
#pragma unroll
        for (int r = 0; r < 8; ++r) {
            float t = fmaxf(s0[r], s1[r]);
#pragma unroll
            for (int off = 1; off < 16; off <<= 1)
                t = fmaxf(t, __shfl_xor(t, off, 32));
            const float nm = fmaxf(mrow[r], t);
            const float corr = __expf(mrow[r] - nm);
            mrow[r] = nm;
            const float p0 = __expf(s0[r] - nm);
            const float p1 = __expf(s1[r] - nm);
            s0[r] = p0; s1[r] = p1;
            float rs = p0 + p1;
#pragma unroll
            for (int off = 1; off < 16; off <<= 1)
                rs += __shfl_xor(rs, off, 32);
            lrow[r] = lrow[r] * corr + rs;
#pragma unroll
            for (int t8 = 0; t8 < 8; ++t8) o[t8][r] *= corr;
        }

        // ---- P: C-layout -> A-layout via per-wave LDS scratch ----
        __bf16* pb = &sP[wave][0];
#pragma unroll
        for (int r = 0; r < 8; ++r) {
            const int m = hi * 8 + r;
            pb[m * 32 + lm]      = (__bf16)s0[r];
            pb[m * 32 + 16 + lm] = (__bf16)s1[r];
        }
        // intra-wave LDS RAW: DS ops are in-order within a wave
        v16bf pf = load_frag(pb + lm * 32, hi);

        // ---- O += P V (8 feature tiles of 16) ----
#pragma unroll
        for (int nb = 0; nb < 8; ++nb) {
            v16bf vf = load_frag(sVt + (nb * 16 + lm) * VLDS, hi);
            o[nb] = wmma_bf16(pf, vf, o[nb]);
        }
        __syncthreads();
    }

    // ---- normalize and store (bf16, layout (B*S, D)) ----
#pragma unroll
    for (int r = 0; r < 8; ++r) {
        const int qr = qbase + hi * 8 + r;
        const float invl = 1.0f / lrow[r];
        const size_t rowoff = ((size_t)(b * S_ + qr)) * D_ + headoff;
#pragma unroll
        for (int nb = 0; nb < 8; ++nb)
            O[rowoff + nb * 16 + lm] = (__bf16)(o[nb][r] * invl);
    }
}

// ---------------------------------------------------------------------------
// Launch
// ---------------------------------------------------------------------------
extern "C" void kernel_launch(void* const* d_in, const int* in_sizes, int n_in,
                              void* d_out, int out_size, void* d_ws, size_t ws_size,
                              hipStream_t stream) {
    const float* x  = (const float*)d_in[0];
    const float* wq = (const float*)d_in[1];
    const float* wk = (const float*)d_in[2];
    const float* wv = (const float*)d_in[3];
    const float* wo = (const float*)d_in[4];

    constexpr size_t BSD = (size_t)B_ * S_ * D_;   // 8388608
    constexpr size_t DD  = (size_t)D_ * D_;        // 4194304

    char* ws = (char*)d_ws;
    __bf16* xb  = (__bf16*)ws;  ws += BSD * 2;
    __bf16* wqb = (__bf16*)ws;  ws += DD * 2;
    __bf16* wkb = (__bf16*)ws;  ws += DD * 2;
    __bf16* wvb = (__bf16*)ws;  ws += DD * 2;
    __bf16* wob = (__bf16*)ws;  ws += DD * 2;
    __bf16* qb  = (__bf16*)ws;  ws += BSD * 2;
    __bf16* kb  = (__bf16*)ws;  ws += BSD * 2;
    __bf16* vb  = (__bf16*)ws;  ws += BSD * 2;
    __bf16* ab  = (__bf16*)ws;  ws += BSD * 2;

    // 1. convert inputs to bf16 (8 elems / thread)
    cvt_f32_bf16<<<2048, 256, 0, stream>>>(x,  xb,  BSD / 8);
    cvt_f32_bf16<<<1024, 256, 0, stream>>>(wq, wqb, DD / 8);
    cvt_f32_bf16<<<1024, 256, 0, stream>>>(wk, wkb, DD / 8);
    cvt_f32_bf16<<<1024, 256, 0, stream>>>(wv, wvb, DD / 8);
    cvt_f32_bf16<<<1024, 256, 0, stream>>>(wo, wob, DD / 8);

    // 2. QKV projections: (B*S, D) = x @ W^T
    dim3 gdim(D_ / 128, (B_ * S_) / 128);  // (16, 32)
    gemm_bf16_nt<true><<<gdim, 256, 0, stream>>>(xb, wqb, nullptr, qb,
                                                 B_ * S_, D_, D_);
    gemm_bf16_nt<true><<<gdim, 256, 0, stream>>>(xb, wkb, nullptr, kb,
                                                 B_ * S_, D_, D_);
    gemm_bf16_nt<true><<<gdim, 256, 0, stream>>>(xb, wvb, nullptr, vb,
                                                 B_ * S_, D_, D_);

    // 3. RoPE on q, k (one thread per rotation pair)
    rope_kernel<<<(B_ * S_ * H_ * 64) / 256, 256, 0, stream>>>(qb, kb);

    // 4. causal flash attention
    flash_kernel<<<B_ * H_ * (S_ / 64), 128, 0, stream>>>(qb, kb, vb, ab);

    // 5. output projection -> f32 d_out
    gemm_bf16_nt<false><<<gdim, 256, 0, stream>>>(ab, wob, (float*)d_out, nullptr,
                                                  B_ * S_, D_, D_);
}
